// PatchLayer_53764400611702
// MI455X (gfx1250) — matrix operand order repaired and checked by reading, not compiled, not run on hardware
//
#include <hip/hip_runtime.h>
#include <hip/hip_bf16.h>

#define IMG    224
#define NPS    14
#define NPATCH 196
#define PDIM   3840     // 16*16*15
#define PROJ   768
#define BATCH  64
#define NTOK   (BATCH * NPATCH)   // 12544
#define LNEPS  1e-6f
#define LDST   40       // LDS row stride in bf16 (32 + 8 pad)
#define KTILES (PDIM / 32)        // 120

typedef __attribute__((ext_vector_type(16))) __bf16 bf16x16;
typedef __attribute__((ext_vector_type(8)))  __bf16 bf16x8;
typedef __attribute__((ext_vector_type(8)))  float  f32x8;

// ---------------------------------------------------------------------------
// Kernel 1: shifted-patch gather + raw patch output + LayerNorm -> bf16 A rows
// One block (256 threads = 8 wave32) per token; 15 elements per thread.
// ---------------------------------------------------------------------------
__global__ __launch_bounds__(256) void k_patch_ln(
    const float* __restrict__ images, const float* __restrict__ gamma,
    const float* __restrict__ beta, float* __restrict__ patches_out,
    __bf16* __restrict__ A)
{
  const int token = blockIdx.x;
  const int t     = threadIdx.x;
  const int bimg  = token / NPATCH;
  const int pq    = token - bimg * NPATCH;
  const int p     = pq / NPS;
  const int q     = pq - p * NPS;

  // concat order: orig, shift(+8,+8), shift(-8,+8), shift(+8,-8), shift(-8,-8)
  const int dy[5] = {0,  8, -8,  8, -8};
  const int dx[5] = {0,  8,  8, -8, -8};

  float vals[15];
  float s1 = 0.f, s2 = 0.f;
#pragma unroll
  for (int jj = 0; jj < 15; ++jj) {
    const int d   = jj * 256 + t;          // 0..3839
    const int r   = d / 240;               // row in patch
    const int rem = d - r * 240;
    const int s   = rem / 15;              // col in patch
    const int c5  = rem - s * 15;          // concat channel
    const int g   = c5 / 3;                // shift group
    const int c   = c5 - g * 3;            // rgb channel
    const int ii  = p * 16 + r + dy[g];
    const int jx  = q * 16 + s + dx[g];
    float v = 0.f;
    if (ii >= 0 && ii < IMG && jx >= 0 && jx < IMG)
      v = images[((size_t)(bimg * IMG + ii) * IMG + jx) * 3 + c];
    vals[jj] = v;
    s1 += v;
    s2 += v * v;
    patches_out[(size_t)token * PDIM + d] = v;
  }

  // wave32 reduction, then cross-wave via LDS
#pragma unroll
  for (int off = 16; off > 0; off >>= 1) {
    s1 += __shfl_xor(s1, off, 32);
    s2 += __shfl_xor(s2, off, 32);
  }
  __shared__ float red[16];
  __shared__ float stats[2];
  const int wave = t >> 5, lane = t & 31;
  if (lane == 0) { red[wave] = s1; red[8 + wave] = s2; }
  __syncthreads();
  if (t == 0) {
    float a = 0.f, b2 = 0.f;
#pragma unroll
    for (int w = 0; w < 8; ++w) { a += red[w]; b2 += red[8 + w]; }
    const float mean = a * (1.f / PDIM);
    const float var  = b2 * (1.f / PDIM) - mean * mean;
    stats[0] = mean;
    stats[1] = rsqrtf(var + LNEPS);
  }
  __syncthreads();
  const float mean = stats[0], inv = stats[1];
#pragma unroll
  for (int jj = 0; jj < 15; ++jj) {
    const int d = jj * 256 + t;
    const float nrm = (vals[jj] - mean) * inv * gamma[d] + beta[d];
    A[(size_t)token * PDIM + d] = (__bf16)nrm;
  }
}

// ---------------------------------------------------------------------------
// Kernel 2: W (3840 x 768 f32, row-major) -> Wt (768 x 3840 bf16) transpose
// ---------------------------------------------------------------------------
__global__ __launch_bounds__(256) void k_wt(
    const float* __restrict__ W, __bf16* __restrict__ Wt)
{
  __shared__ float tile[32][33];
  const int k0 = blockIdx.x * 32;   // along K = 3840
  const int n0 = blockIdx.y * 32;   // along N = 768
  const int tx = threadIdx.x & 31;
  const int ty = threadIdx.x >> 5;  // 0..7
#pragma unroll
  for (int i = 0; i < 32; i += 8)
    tile[ty + i][tx] = W[(size_t)(k0 + ty + i) * PROJ + (n0 + tx)];
  __syncthreads();
#pragma unroll
  for (int i = 0; i < 32; i += 8)
    Wt[(size_t)(n0 + ty + i) * PDIM + (k0 + tx)] = (__bf16)tile[tx][ty + i];
}

// ---------------------------------------------------------------------------
// Kernel 3: tokens = A(12544x3840 bf16) * Wt^T + bias.
// Double-buffered async-to-LDS pipeline (ASYNCcnt), 1 barrier per K-step.
// Block tile 128x128, 8 waves, wave tile 32x64 (2x4 fragments), K step 32.
// ---------------------------------------------------------------------------
#define ASYNC_CP_B128(lds32, gaddr64)                                   \
  asm volatile("global_load_async_to_lds_b128 %0, %1, off"              \
               :: "v"(lds32), "v"(gaddr64) : "memory")

__global__ __launch_bounds__(256) void k_gemm(
    const __bf16* __restrict__ A, const __bf16* __restrict__ Wt,
    const float* __restrict__ bias, float* __restrict__ C)
{
  __shared__ __bf16 As[2][128 * LDST];   // 2 x 10240 B
  __shared__ __bf16 Bs[2][128 * LDST];   // 2 x 10240 B

  const int t    = threadIdx.x;
  const int wave = t >> 5;
  const int lane = t & 31;
  const int wm   = wave & 3;          // 4 waves along M
  const int wn   = wave >> 2;         // 2 waves along N
  const int l15  = lane & 15;
  const int lhi  = lane >> 4;
  const int M0   = blockIdx.y * 128;
  const int N0   = blockIdx.x * 128;

  // copy plan: 512 x 16B chunks per tile; thread t owns rows r0 and r0+64
  const int r0 = t >> 2;
  const int k0 = (t & 3) * 8;

  const unsigned long long gA0 =
      (unsigned long long)(uintptr_t)(A + (size_t)(M0 + r0) * PDIM + k0);
  const unsigned long long gA1 = gA0 + (unsigned long long)64 * PDIM * 2;
  const unsigned long long gB0 =
      (unsigned long long)(uintptr_t)(Wt + (size_t)(N0 + r0) * PDIM + k0);
  const unsigned long long gB1 = gB0 + (unsigned long long)64 * PDIM * 2;

  const unsigned lA0 = (unsigned)(uintptr_t)&As[0][r0 * LDST + k0];
  const unsigned lA1 = lA0 + 64 * LDST * 2;
  const unsigned lB0 = (unsigned)(uintptr_t)&Bs[0][r0 * LDST + k0];
  const unsigned lB1 = lB0 + 64 * LDST * 2;
  const unsigned bufB = 128 * LDST * 2;   // 10240 B between buffers

  // accumulators initialized with bias (all 8 C VGPRs of a lane share column l15)
  f32x8 acc[2][4];
#pragma unroll
  for (int j = 0; j < 4; ++j) {
    const float bv = bias[N0 + wn * 64 + j * 16 + l15];
#pragma unroll
    for (int i = 0; i < 2; ++i)
#pragma unroll
      for (int e = 0; e < 8; ++e) acc[i][j][e] = bv;
  }

  // issue tile kt into buffer b (4 x b128 per thread, no VGPR staging)
  auto issue = [&](int kt, int b) {
    const unsigned long long go = (unsigned long long)kt * 64;  // 32 bf16 = 64 B
    const unsigned bo = b ? bufB : 0u;
    ASYNC_CP_B128(lA0 + bo, gA0 + go);
    ASYNC_CP_B128(lA1 + bo, gA1 + go);
    ASYNC_CP_B128(lB0 + bo, gB0 + go);
    ASYNC_CP_B128(lB1 + bo, gB1 + go);
  };

  issue(0, 0);
  int buf = 0;

  for (int kt = 0; kt < KTILES; ++kt) {
    asm volatile("s_wait_asynccnt 0x0" ::: "memory");  // my writes for tile kt done
    __syncthreads();                                    // everyone's writes visible
    if (kt < KTILES - 1) issue(kt + 1, buf ^ 1);        // prefetch overlaps WMMA

    const __bf16* Asb = As[buf];
    const __bf16* Bsb = Bs[buf];

    // A fragments: lane<16 -> M=l15, K in [0..7]+[16..23]; lane>=16 -> K+8
    bf16x16 af[2];
#pragma unroll
    for (int i = 0; i < 2; ++i) {
      const __bf16* base = &Asb[(wm * 32 + i * 16 + l15) * LDST + lhi * 8];
      bf16x8 lo = *(const bf16x8*)base;
      bf16x8 hi = *(const bf16x8*)(base + 16);
      af[i] = __builtin_shufflevector(lo, hi, 0,1,2,3,4,5,6,7,8,9,10,11,12,13,14,15);
    }
    // B fragments: lane<16 -> N=l15, K in [0..7]+[16..23]; lane>=16 -> K+8
    bf16x16 bfm[4];
#pragma unroll
    for (int j = 0; j < 4; ++j) {
      const __bf16* base = &Bsb[(wn * 64 + j * 16 + l15) * LDST + lhi * 8];
      bf16x8 lo = *(const bf16x8*)base;
      bf16x8 hi = *(const bf16x8*)(base + 16);
      bfm[j] = __builtin_shufflevector(lo, hi, 0,1,2,3,4,5,6,7,8,9,10,11,12,13,14,15);
    }

#pragma unroll
    for (int i = 0; i < 2; ++i)
#pragma unroll
      for (int j = 0; j < 4; ++j)
        acc[i][j] = __builtin_amdgcn_wmma_f32_16x16x32_bf16(
            false, af[i], false, bfm[j], (short)0, acc[i][j], false, false);

    buf ^= 1;
  }

  // epilogue: C VGPR e -> row = e + 8*lhi, col = l15
#pragma unroll
  for (int i = 0; i < 2; ++i)
#pragma unroll
    for (int j = 0; j < 4; ++j) {
      const int col = N0 + wn * 64 + j * 16 + l15;
#pragma unroll
      for (int e = 0; e < 8; ++e) {
        const int row = M0 + wm * 32 + i * 16 + lhi * 8 + e;
        C[(size_t)row * PROJ + col] = acc[i][j][e];
      }
    }
}

// ---------------------------------------------------------------------------
extern "C" void kernel_launch(void* const* d_in, const int* in_sizes, int n_in,
                              void* d_out, int out_size, void* d_ws, size_t ws_size,
                              hipStream_t stream) {
  const float* images = (const float*)d_in[0];
  const float* gamma  = (const float*)d_in[1];
  const float* beta   = (const float*)d_in[2];
  const float* W      = (const float*)d_in[3];
  const float* bias   = (const float*)d_in[4];

  float* tokens  = (float*)d_out;                       // 12544 x 768
  float* patches = tokens + (size_t)NTOK * PROJ;        // 12544 x 3840

  __bf16* Abf = (__bf16*)d_ws;                                          // 96.3 MB
  __bf16* Wt  = (__bf16*)((char*)d_ws + (size_t)NTOK * PDIM * 2);       // 5.9 MB

  k_patch_ln<<<NTOK, 256, 0, stream>>>(images, gamma, beta, patches, Abf);
  k_wt<<<dim3(PDIM / 32, PROJ / 32), 256, 0, stream>>>(W, Wt);
  k_gemm<<<dim3(PROJ / 128, NTOK / 128), 256, 0, stream>>>(Abf, Wt, bias, tokens);
}
